// TwoSimplicialAttention_24515673325872
// MI455X (gfx1250) — matrix operand order repaired and checked by reading, not compiled, not run on hardware
//
#include <hip/hip_runtime.h>
#include <hip/hip_bf16.h>

typedef __attribute__((ext_vector_type(16))) __bf16 v16bf;
typedef __attribute__((ext_vector_type(8)))  __bf16 v8bf;
typedef __attribute__((ext_vector_type(4)))  __bf16 v4bf;
typedef __attribute__((ext_vector_type(8)))  float  v8f;

#define NHEAD 8
#define DDIM  64
#define WWIN  64
#define LDB   72   // bf16 LDS row stride (elems): 144 B, 16B-aligned
#define LDV   64   // f32 LDS row stride for v1 (256 B rows, b128-aligned)
#define NWAVE 4
#define QKSCALE 0.125f

union FragU { v16bf v; v8bf h[2]; };

// A-operand of v_wmma_f32_16x16x32_bf16 (16x32, M=lane%16):
// lane half lh: chunk0 = K[kb*32 + 8*lh .. +7], chunk1 = +16
__device__ __forceinline__ v16bf frag_A(const __bf16* base, int row, int kb32, int lh) {
  FragU f;
  const __bf16* p = base + row * LDB + kb32 + 8 * lh;
  f.h[0] = *(const v8bf*)(p);
  f.h[1] = *(const v8bf*)(p + 16);
  return f.v;
}
// B-operand (32x16, N=lane%16): chunk0 = K[kb*32 + 8*lh .. +7], chunk1 = +16
__device__ __forceinline__ v16bf frag_B(const __bf16* base, int nrow, int kb32, int lh) {
  FragU f;
  const __bf16* p = base + nrow * LDB + kb32 + 8 * lh;
  f.h[0] = *(const v8bf*)(p);
  f.h[1] = *(const v8bf*)(p + 16);
  return f.v;
}

__global__ __launch_bounds__(128, 1)
void twosimp_attn_kernel(const float* __restrict__ q,
                         const float* __restrict__ k1,
                         const float* __restrict__ k2,
                         const float* __restrict__ v1,
                         const float* __restrict__ v2,
                         float* __restrict__ out)
{
  __shared__ __bf16 Abf[WWIN * LDB];   // bf16(q * k1w)         [a][d]
  __shared__ __bf16 Bbf[WWIN * LDB];   // bf16(k2w + 1)         [c][d]
  __shared__ __bf16 V2T[DDIM * LDB];   // bf16(v2w + 1)         [d][c]  (B-op of GEMM2)
  __shared__ __bf16 Pbf[WWIN * LDB];   // softmax probs bf16    [a][c]  (A-op of GEMM2)
  __shared__ float  v1s[WWIN * LDV];   // v1w f32               [a][d]  (async-copied)
  __shared__ float  qs[DDIM];
  __shared__ float  redmax[NWAVE];
  __shared__ float  redsum[NWAVE];
  __shared__ float  obuf[NWAVE * DDIM];

  const int tid   = threadIdx.x;
  const int lane  = tid & 31;
  const int wave  = tid >> 5;
  const int lh    = lane >> 4;   // 0/1: half selector
  const int lm    = lane & 15;   // N (or M for A-op) within tile

  const int s = blockIdx.x >> 3;
  const int h = blockIdx.x & 7;
  const long rowBase = (long)s * (NHEAD * DDIM) + h * DDIM;

  if (tid < DDIM) qs[tid] = q[rowBase + tid];

  // ---- Phase 1a: fire async LDS copy of the v1 window (pure copy -> TDM-free
  // async path, tracked by ASYNCcnt; overlaps with the staging arithmetic) ----
#pragma unroll
  for (int i = 0; i < (WWIN * DDIM / 4) / 128; ++i) {
    const int idx = tid + i * 128;
    const int a = idx >> 4;                  // window row 0..63
    const int d = (idx & 15) * 4;            // 16B chunk
    const int t = (s - a) > 0 ? (s - a) : 0; // clamp (p==0 there anyway)
    const float* gp = v1 + (long)t * (NHEAD * DDIM) + h * DDIM + d;
    const unsigned lp = (unsigned)(unsigned long long)(&v1s[a * LDV + d]);
    asm volatile("global_load_async_to_lds_b128 %0, %1, off"
                 :: "v"(lp), "v"(gp) : "memory");
  }
  __syncthreads();   // qs visible

  // ---- Phase 1b: stage WMMA operands (vectorized b128 loads, bf16 packing) ----
#pragma unroll
  for (int i = 0; i < (WWIN * DDIM / 4) / 128; ++i) {
    const int idx = tid + i * 128;
    const int a = idx >> 4;
    const int d = (idx & 15) * 4;
    const int t = (s - a) > 0 ? (s - a) : 0;
    const long g = (long)t * (NHEAD * DDIM) + h * DDIM + d;
    const float4 kk1 = *(const float4*)(k1 + g);
    const float4 kk2 = *(const float4*)(k2 + g);
    const float4 vv2 = *(const float4*)(v2 + g);
    const float4 qq  = *(const float4*)(&qs[d]);
    v4bf av, bv;
    av[0] = (__bf16)(qq.x * kk1.x);  av[1] = (__bf16)(qq.y * kk1.y);
    av[2] = (__bf16)(qq.z * kk1.z);  av[3] = (__bf16)(qq.w * kk1.w);
    bv[0] = (__bf16)(kk2.x + 1.0f);  bv[1] = (__bf16)(kk2.y + 1.0f);
    bv[2] = (__bf16)(kk2.z + 1.0f);  bv[3] = (__bf16)(kk2.w + 1.0f);
    *(v4bf*)(Abf + a * LDB + d) = av;       // ds_store_b64
    *(v4bf*)(Bbf + a * LDB + d) = bv;       // ds_store_b64
    V2T[(d + 0) * LDB + a] = (__bf16)(vv2.x + 1.0f);   // transposed scatter
    V2T[(d + 1) * LDB + a] = (__bf16)(vv2.y + 1.0f);
    V2T[(d + 2) * LDB + a] = (__bf16)(vv2.z + 1.0f);
    V2T[(d + 3) * LDB + a] = (__bf16)(vv2.w + 1.0f);
  }
  asm volatile("s_wait_asynccnt 0" ::: "memory");  // v1s copy landed
  __syncthreads();

  // ---- Phase 2: GEMM1  logits[a, c] (each wave owns 16 a-rows) ----
  const int mBase = wave * 16;
  v8f acc[4];
#pragma unroll
  for (int tc = 0; tc < 4; ++tc)
#pragma unroll
    for (int r = 0; r < 8; ++r) acc[tc][r] = 0.0f;

#pragma unroll
  for (int kb = 0; kb < 2; ++kb) {
    v16bf af = frag_A(Abf, mBase + lm, kb * 32, lh);
#pragma unroll
    for (int tc = 0; tc < 4; ++tc) {
      v16bf bf = frag_B(Bbf, tc * 16 + lm, kb * 32, lh);
      acc[tc] = __builtin_amdgcn_wmma_f32_16x16x32_bf16(
          false, af, false, bf, (short)0, acc[tc], false, false);
    }
  }

  // ---- Phase 3: scale + causal-window mask + joint softmax over 64x64 ----
  float lmax = -__builtin_inff();
#pragma unroll
  for (int tc = 0; tc < 4; ++tc) {
#pragma unroll
    for (int r = 0; r < 8; ++r) {
      const int a = mBase + r + 8 * lh;    // C/D layout: M = r + 8*(lane/16)
      const int c = tc * 16 + lm;          //             N = lane%16
      float x = acc[tc][r] * QKSCALE;
      if (a > s || c > s) x = -__builtin_inff();
      acc[tc][r] = x;
      lmax = fmaxf(lmax, x);
    }
  }
#pragma unroll
  for (int off = 16; off >= 1; off >>= 1)
    lmax = fmaxf(lmax, __shfl_xor(lmax, off, 32));
  if (lane == 0) redmax[wave] = lmax;
  __syncthreads();
  const float gmax = fmaxf(fmaxf(redmax[0], redmax[1]), fmaxf(redmax[2], redmax[3]));

  float lsum = 0.0f;
#pragma unroll
  for (int tc = 0; tc < 4; ++tc)
#pragma unroll
    for (int r = 0; r < 8; ++r) {
      const float e = __expf(acc[tc][r] - gmax);
      acc[tc][r] = e;
      lsum += e;
    }
#pragma unroll
  for (int off = 16; off >= 1; off >>= 1)
    lsum += __shfl_xor(lsum, off, 32);
  if (lane == 0) redsum[wave] = lsum;
  __syncthreads();
  const float inv = 1.0f / (redsum[0] + redsum[1] + redsum[2] + redsum[3]);

  // D-fragment -> A-operand layout via LDS (wave touches only its own rows;
  // per-wave DS ops are in-order so no extra barrier needed)
#pragma unroll
  for (int tc = 0; tc < 4; ++tc)
#pragma unroll
    for (int r = 0; r < 8; ++r)
      Pbf[(mBase + r + 8 * lh) * LDB + tc * 16 + lm] = (__bf16)(acc[tc][r] * inv);

  // ---- Phase 4: GEMM2  tmp[a, d] = P @ (v2w+1) ----
  v8f t2[4];
#pragma unroll
  for (int td = 0; td < 4; ++td)
#pragma unroll
    for (int r = 0; r < 8; ++r) t2[td][r] = 0.0f;

#pragma unroll
  for (int kb = 0; kb < 2; ++kb) {
    v16bf pf = frag_A(Pbf, mBase + lm, kb * 32, lh);
#pragma unroll
    for (int td = 0; td < 4; ++td) {
      v16bf vf = frag_B(V2T, td * 16 + lm, kb * 32, lh);
      t2[td] = __builtin_amdgcn_wmma_f32_16x16x32_bf16(
          false, pf, false, vf, (short)0, t2[td], false, false);
    }
  }

  // ---- Phase 5: out[d] = sum_a tmp[a,d] * v1w[a,d] ----
#pragma unroll
  for (int td = 0; td < 4; ++td) {
    float sum = 0.0f;
#pragma unroll
    for (int r = 0; r < 8; ++r) {
      const int a = mBase + r + 8 * lh;
      const int d = td * 16 + lm;
      sum += t2[td][r] * v1s[a * LDV + d];
    }
    sum += __shfl_xor(sum, 16, 32);        // fold the two M-halves (same d)
    if (lh == 0) obuf[wave * DDIM + td * 16 + lm] = sum;
  }
  __syncthreads();

  if (tid < DDIM) {
    out[rowBase + tid] = obuf[tid] + obuf[DDIM + tid] +
                         obuf[2 * DDIM + tid] + obuf[3 * DDIM + tid];
  }
}

extern "C" void kernel_launch(void* const* d_in, const int* in_sizes, int n_in,
                              void* d_out, int out_size, void* d_ws, size_t ws_size,
                              hipStream_t stream) {
  const float* q  = (const float*)d_in[0];
  const float* k1 = (const float*)d_in[1];
  const float* k2 = (const float*)d_in[2];
  const float* v1 = (const float*)d_in[3];
  const float* v2 = (const float*)d_in[4];
  float* out = (float*)d_out;

  const int S = in_sizes[0] / (NHEAD * DDIM);   // 1024
  dim3 grid(S * NHEAD);                         // one block per (s, h)
  dim3 block(128);                              // 4 waves (wave32)
  twosimp_attn_kernel<<<grid, block, 0, stream>>>(q, k1, k2, v1, v2, out);
}
